// LongTermMemory_79413945303744
// MI455X (gfx1250) — matrix operand order repaired and checked by reading, not compiled, not run on hardware
//
#include <hip/hip_runtime.h>
#include <hip/hip_bf16.h>
#include <math.h>

// ---------------------------------------------------------------------------
// Problem sizes (fixed by the reference):
//   x:   [4,2048,1024]  -> R=8192 rows, C=1024
//   ltm: [8192,1024]    -> M=8192
//   W1:  [1024,2048], b1:[2048], W2:[2048,1024], b2:[1024], k=16
// ---------------------------------------------------------------------------

typedef __attribute__((ext_vector_type(16))) __bf16 v16bf;
typedef __attribute__((ext_vector_type(8)))  __bf16 v8bf;
typedef __attribute__((ext_vector_type(8)))  float  v8f;

__device__ __forceinline__ v16bf cat16(v8bf lo, v8bf hi) {
  return __builtin_shufflevector(lo, hi, 0, 1, 2, 3, 4, 5, 6, 7,
                                         8, 9, 10, 11, 12, 13, 14, 15);
}

__device__ __forceinline__ v8f wmma_bf16(v16bf a, v16bf b, v8f c) {
  // (neg_a, A, neg_b, B, c_mod, C, reuse_a, reuse_b)
  return __builtin_amdgcn_wmma_f32_16x16x32_bf16(false, a, false, b,
                                                 (short)0, c, false, false);
}

// ---------------------------------------------------------------------------
// Tensor Data Mover: 2D tile load (Global -> LDS), bf16 elements.
// D# layout per CDNA5 ISA ch.8: group0 = {count, lds_addr, global_addr, type=2},
// group1 = {data_size=2B, tensor_dim0/1, tile_dim0/1, tensor_dim0_stride}.
// Issued by one wave; completion via s_wait_tensorcnt; visibility to the rest
// of the workgroup via the barrier that follows the wait.
// ---------------------------------------------------------------------------
#if defined(__has_builtin)
#if __has_builtin(__builtin_amdgcn_tensor_load_to_lds)
#define LTM_USE_TDM 1
#endif
#endif
#ifndef LTM_USE_TDM
#define LTM_USE_TDM 0
#endif

#if LTM_USE_TDM
typedef __attribute__((ext_vector_type(4))) unsigned int v4u;
typedef __attribute__((ext_vector_type(8))) int v8i_t;
typedef __attribute__((ext_vector_type(4))) int v4i_t;

__device__ __forceinline__ void tdm_load_2d_bf16(unsigned int ldsOff,
                                                 const void* gaddr,
                                                 unsigned int tensorD0,
                                                 unsigned int tensorD1,
                                                 unsigned int tileD0,
                                                 unsigned int tileD1,
                                                 unsigned int strideD0) {
  const unsigned long long ga = (unsigned long long)(size_t)gaddr;
  v4u g0;
  g0[0] = 1u;                                       // count=1 (user descriptor)
  g0[1] = ldsOff;                                   // lds_addr (bytes)
  g0[2] = (unsigned int)(ga & 0xffffffffu);         // global_addr[31:0]
  g0[3] = (unsigned int)((ga >> 32) & 0x01ffffffu)  // global_addr[56:32]
          | 0x80000000u;                            // type=2 ("image")
  v8i_t g1;
  g1[0] = (int)(1u << 16);                          // data_size=1 (2 bytes)
  g1[1] = (int)((tensorD0 & 0xffffu) << 16);        // tensor_dim0[15:0]
  g1[2] = (int)(((tensorD0 >> 16) & 0xffffu) |      // tensor_dim0[31:16]
                ((tensorD1 & 0xffffu) << 16));      // tensor_dim1[15:0]
  g1[3] = (int)(((tensorD1 >> 16) & 0xffffu) |      // tensor_dim1[31:16]
                ((tileD0 & 0xffffu) << 16));        // tile_dim0
  g1[4] = (int)(tileD1 & 0xffffu);                  // tile_dim1 (tile_dim2=0)
  g1[5] = (int)strideD0;                            // tensor_dim0_stride[31:0]
  g1[6] = 0;                                        // stride hi / dim1_stride
  g1[7] = 0;
  v4i_t gz = {0, 0, 0, 0};
#if __clang_major__ >= 23
  v8i_t gz8 = {0, 0, 0, 0, 0, 0, 0, 0};
  __builtin_amdgcn_tensor_load_to_lds(g0, g1, gz, gz, gz8, 0);
#else
  __builtin_amdgcn_tensor_load_to_lds(g0, g1, gz, gz, 0);
#endif
}
#endif  // LTM_USE_TDM

// ---------------------------------------------------------------------------
// Row L2-norm (inverse, clamped at eps) + fp32 -> bf16 conversion.
// ---------------------------------------------------------------------------
__global__ void __launch_bounds__(256)
rownorm_cvt_kernel(const float* __restrict__ in, __bf16* __restrict__ outb,
                   float* __restrict__ inv, int C) {
  __shared__ float red[256];
  const int row = blockIdx.x;
  const float* r = in + (size_t)row * C;
  float ss = 0.f;
  for (int i = threadIdx.x; i < C; i += 256) {
    float v = r[i];
    ss += v * v;
    outb[(size_t)row * C + i] = (__bf16)v;
  }
  red[threadIdx.x] = ss;
  __syncthreads();
  for (int s = 128; s > 0; s >>= 1) {
    if (threadIdx.x < s) red[threadIdx.x] += red[threadIdx.x + s];
    __syncthreads();
  }
  if (threadIdx.x == 0) inv[row] = 1.f / fmaxf(sqrtf(red[0]), 1e-12f);
}

// Plain fp32 -> bf16 (weights).
__global__ void __launch_bounds__(256)
cvt_bf16_kernel(const float* __restrict__ in, __bf16* __restrict__ out, int n) {
  int i = blockIdx.x * 256 + threadIdx.x;
  const int stride = gridDim.x * 256;
  for (; i < n; i += stride) out[i] = (__bf16)in[i];
}

// ---------------------------------------------------------------------------
// Tiled bf16 WMMA GEMM: C[M,N] = epilogue(A[M,K] @ B[K,N] + bias[N])
// Block tile 128x128, K-step 32, 8 waves each computing a 64x32 sub-tile.
//   EPI==0: exact GELU, store bf16;  EPI==1: bias only, store fp32.
// ---------------------------------------------------------------------------
template <int EPI>
__global__ void __launch_bounds__(256)
gemm_bf16_wmma(const __bf16* __restrict__ A, const __bf16* __restrict__ B,
               const float* __restrict__ bias, void* __restrict__ outv,
               int M, int N, int K) {
  __shared__ __align__(16) __bf16 ldsA[128 * 32];   // [m][k]
  __shared__ __align__(16) __bf16 ldsBt[128 * 32];  // transposed: [n][k]

  const int tid  = threadIdx.x;
  const int wave = tid >> 5, lane = tid & 31;
  const int wm = wave >> 2, wn = wave & 3;  // 2 (M) x 4 (N) wave grid
  const int l16 = lane & 15, sub = lane >> 4;
  const int rowBase = blockIdx.y * 128;
  const int colBase = blockIdx.x * 128;

  v8f zero = {};
  v8f acc[4][2];
#pragma unroll
  for (int mi = 0; mi < 4; ++mi)
#pragma unroll
    for (int ni = 0; ni < 2; ++ni) acc[mi][ni] = zero;

  for (int kt = 0; kt < K; kt += 32) {
    for (int idx = tid; idx < 512; idx += 256) {
      const int m = idx >> 2, k8 = (idx & 3) << 3;
      *(v8bf*)(ldsA + m * 32 + k8) =
          *(const v8bf*)(A + (size_t)(rowBase + m) * K + kt + k8);
    }
    for (int idx = tid; idx < 512; idx += 256) {
      const int k = idx >> 4, n8 = (idx & 15) << 3;
      v8bf v = *(const v8bf*)(B + (size_t)(kt + k) * N + colBase + n8);
#pragma unroll
      for (int j = 0; j < 8; ++j) ldsBt[(n8 + j) * 32 + k] = v[j];
    }
    __syncthreads();

    v16bf afr[4], bfr[2];
#pragma unroll
    for (int mi = 0; mi < 4; ++mi) {
      const int m = wm * 64 + mi * 16 + l16;
      afr[mi] = cat16(*(const v8bf*)(ldsA + m * 32 + 8 * sub),
                      *(const v8bf*)(ldsA + m * 32 + 16 + 8 * sub));
    }
#pragma unroll
    for (int ni = 0; ni < 2; ++ni) {
      const int n = wn * 32 + ni * 16 + l16;
      bfr[ni] = cat16(*(const v8bf*)(ldsBt + n * 32 + 8 * sub),
                      *(const v8bf*)(ldsBt + n * 32 + 16 + 8 * sub));
    }
#pragma unroll
    for (int mi = 0; mi < 4; ++mi)
#pragma unroll
      for (int ni = 0; ni < 2; ++ni)
        acc[mi][ni] = wmma_bf16(afr[mi], bfr[ni], acc[mi][ni]);
    __syncthreads();
  }

#pragma unroll
  for (int mi = 0; mi < 4; ++mi) {
#pragma unroll
    for (int ni = 0; ni < 2; ++ni) {
      const int col = colBase + wn * 32 + ni * 16 + l16;
      const float bv = bias[col];
#pragma unroll
      for (int v = 0; v < 8; ++v) {
        const int row = rowBase + wm * 64 + mi * 16 + v + 8 * sub;
        float r = acc[mi][ni][v] + bv;
        if (EPI == 0) {
          r = 0.5f * r * (1.f + erff(r * 0.70710678118654752f));
          ((__bf16*)outv)[(size_t)row * N + col] = (__bf16)r;
        } else {
          ((float*)outv)[(size_t)row * N + col] = r;
        }
      }
    }
  }
}

// ---------------------------------------------------------------------------
// Fused cosine-score GEMM + top-16 + softmax + weighted gather.
//
// Workgroup = 8 waves, owns 128 query rows. LDS layout (296 KB of 320 KB):
//   [0,       256K) : 128x1024 bf16 query block (pinned for whole kernel)
//   [256K,    272K) : double-buffered B k-slab, 2 x (128 cols x 32 k) bf16
//   [272K..296K)    : per-wave 16x16 score tile + (w,idx) scratch
//
// Outer loop over 64 col-blocks of 128 buffer rows; inner K loop stages the
// shared B slab once per workgroup (TDM async tensor load, double-buffered,
// s_wait_tensorcnt 1 pipelining), then every wave does 8 WMMAs per A-fragment
// (16x128 wave tile, 64 acc VGPRs). Scores never touch HBM.
// ---------------------------------------------------------------------------
#define SC_BT_OFF   262144u                 // Bt buffers (2 x 8192 B)
#define SC_S_OFF    (262144u + 16384u)      // 8 waves * 16*16 f32
#define SC_W_OFF    (SC_S_OFF + 8192u)      // 8 waves * 16*16 f32
#define SC_I_OFF    (SC_W_OFF + 8192u)      // 8 waves * 16*16 i32
#define SC_LDS_SIZE (SC_I_OFF + 8192u)      // 303104 B

__global__ void __launch_bounds__(256)
scores_topk_kernel(const __bf16* __restrict__ xb, const __bf16* __restrict__ mb,
                   const float* __restrict__ xinv, const float* __restrict__ minv,
                   const float* __restrict__ ltm, float* __restrict__ out,
                   int Mbuf) {
  extern __shared__ __align__(16) char smem[];
  __bf16* ldsA  = (__bf16*)smem;
  __bf16* ldsBt = (__bf16*)(smem + SC_BT_OFF);  // [buf][128 n][32 k]
  float*  ldsS  = (float*)(smem + SC_S_OFF);
  float*  ldsW  = (float*)(smem + SC_W_OFF);
  int*    ldsI  = (int*)(smem + SC_I_OFF);

  const int tid = threadIdx.x, wave = tid >> 5, lane = tid & 31;
  const int l16 = lane & 15, sub = lane >> 4;
  const int rowBase = blockIdx.x * 128;

  // ---- Stage the 128x1024 query block once (TDM if available). ----
#if LTM_USE_TDM
  if (wave == 0) {
    tdm_load_2d_bf16(0u, xb + (size_t)rowBase * 1024,
                     /*tensorD0=*/1024, /*tensorD1=*/8192,
                     /*tileD0=*/1024, /*tileD1=*/128, /*strideD0=*/1024);
    __builtin_amdgcn_s_wait_tensorcnt(0);
  }
#else
  for (int idx = tid; idx < 16384; idx += 256) {
    const int m = idx >> 7, k8 = (idx & 127) << 3;
    *(v8bf*)(ldsA + m * 1024 + k8) =
        *(const v8bf*)(xb + (size_t)(rowBase + m) * 1024 + k8);
  }
#endif
  __syncthreads();

  const int mLocal = wave * 16;  // this wave's 16 rows (block-local)
  float xi[8];
#pragma unroll
  for (int v = 0; v < 8; ++v) xi[v] = xinv[rowBase + mLocal + v + 8 * sub];

  float topv[16];
  int   topi[16];
#pragma unroll
  for (int j = 0; j < 16; ++j) { topv[j] = -1e30f; topi[j] = 0; }

  float* myS = ldsS + wave * 256;
  float* myW = ldsW + wave * 256;
  int*   myI = ldsI + wave * 256;

  const int nBlocks = Mbuf >> 7;  // col-blocks of 128 buffer rows
  for (int nb = 0; nb < nBlocks; ++nb) {
    const int colBase = nb << 7;

    v8f acc[8];
    {
      v8f zero = {};
#pragma unroll
      for (int ni = 0; ni < 8; ++ni) acc[ni] = zero;
    }

#if LTM_USE_TDM
    // Prologue: async-load the kt=0 slab into buffer 0.
    if (wave == 0)
      tdm_load_2d_bf16(SC_BT_OFF, mb + (size_t)colBase * 1024,
                       1024, 8192, /*tileD0=*/32, /*tileD1=*/128, 1024);
#endif

    for (int kt = 0; kt < 1024; kt += 32) {
      const int buf = (kt >> 5) & 1;
#if LTM_USE_TDM
      if (wave == 0) {
        if (kt + 32 < 1024) {
          // Prefetch next slab into the other buffer, then wait for current.
          tdm_load_2d_bf16(SC_BT_OFF + (unsigned)(buf ^ 1) * 8192u,
                           mb + (size_t)colBase * 1024 + kt + 32,
                           1024, 8192, 32, 128, 1024);
          __builtin_amdgcn_s_wait_tensorcnt(1);
        } else {
          __builtin_amdgcn_s_wait_tensorcnt(0);
        }
      }
#else
      // Manual staging of the shared 128x32 slab (mb rows are k-contiguous,
      // so this is a straight strided copy, no transpose needed).
      for (int idx = tid; idx < 512; idx += 256) {
        const int n = idx >> 2, k8 = (idx & 3) << 3;
        *(v8bf*)(ldsBt + buf * 4096 + n * 32 + k8) =
            *(const v8bf*)(mb + (size_t)(colBase + n) * 1024 + kt + k8);
      }
#endif
      __syncthreads();

      const __bf16* ar = ldsA + (mLocal + l16) * 1024 + kt;
      const v16bf a = cat16(*(const v8bf*)(ar + 8 * sub),
                            *(const v8bf*)(ar + 16 + 8 * sub));
      const __bf16* bt = ldsBt + buf * 4096;
#pragma unroll
      for (int ni = 0; ni < 8; ++ni) {
        const __bf16* br = bt + (ni * 16 + l16) * 32;
        const v16bf b = cat16(*(const v8bf*)(br + 8 * sub),
                              *(const v8bf*)(br + 16 + 8 * sub));
        acc[ni] = wmma_bf16(a, b, acc[ni]);
      }
      __syncthreads();
    }

    // Scale by 1/(||x||*||m||), transpose 16x16 sub-tiles through LDS, and
    // merge into each row's sorted top-16 (lanes 0..15 own one row each).
#pragma unroll 1
    for (int ni = 0; ni < 8; ++ni) {
      const float mi = minv[colBase + ni * 16 + l16];
#pragma unroll
      for (int v = 0; v < 8; ++v)
        myS[(v + 8 * sub) * 16 + l16] = acc[ni][v] * xi[v] * mi;
      if (lane < 16) {
        for (int j = 0; j < 16; ++j) {
          const float s = myS[lane * 16 + j];
          if (s > topv[15]) {
            int p = 15;
            while (p > 0 && topv[p - 1] < s) {
              topv[p] = topv[p - 1];
              topi[p] = topi[p - 1];
              --p;
            }
            topv[p] = s;
            topi[p] = colBase + ni * 16 + j;
          }
        }
      }
    }
  }

  // Softmax over the (sorted, max-first) top-16; publish (w, idx) to LDS.
  if (lane < 16) {
    const float mx = topv[0];
    float e[16], sum = 0.f;
#pragma unroll
    for (int j = 0; j < 16; ++j) { e[j] = expf(topv[j] - mx); sum += e[j]; }
    const float rs = 1.f / sum;
#pragma unroll
    for (int j = 0; j < 16; ++j) {
      myW[lane * 16 + j] = e[j] * rs;
      myI[lane * 16 + j] = topi[j];
    }
  }

  // Whole wave cooperatively gathers the 16 raw fp32 buffer rows per query
  // row and accumulates into mem_out (already in d_out from GEMM2).
  for (int r = 0; r < 16; ++r) {
    const int row = rowBase + mLocal + r;
    float w[16];
    int   id[16];
#pragma unroll
    for (int j = 0; j < 16; ++j) { w[j] = myW[r * 16 + j]; id[j] = myI[r * 16 + j]; }
    for (int c = lane; c < 1024; c += 32) {
      float a = 0.f;
#pragma unroll
      for (int j = 0; j < 16; ++j) a += w[j] * ltm[(size_t)id[j] * 1024 + c];
      out[(size_t)row * 1024 + c] += a;
    }
  }
}

// ---------------------------------------------------------------------------
// Host-side orchestration.
// ---------------------------------------------------------------------------
extern "C" void kernel_launch(void* const* d_in, const int* in_sizes, int n_in,
                              void* d_out, int out_size, void* d_ws,
                              size_t ws_size, hipStream_t stream) {
  (void)in_sizes; (void)n_in; (void)out_size; (void)ws_size;

  const float* x   = (const float*)d_in[0];  // [8192,1024]
  const float* ltm = (const float*)d_in[1];  // [8192,1024]
  const float* W1  = (const float*)d_in[2];  // [1024,2048]
  const float* b1  = (const float*)d_in[3];  // [2048]
  const float* W2  = (const float*)d_in[4];  // [2048,1024]
  const float* b2  = (const float*)d_in[5];  // [1024]
  float* out = (float*)d_out;

  const int R = 8192, C = 1024, H = 2048, M = 8192;

  // Workspace layout (bytes): ~72 MB total.
  char* ws = (char*)d_ws;
  __bf16* xb   = (__bf16*)(ws + 0);                    // 16 MB
  __bf16* mb   = (__bf16*)(ws + (16u << 20));          // 16 MB
  __bf16* w1b  = (__bf16*)(ws + (32u << 20));          // 4 MB
  __bf16* w2b  = (__bf16*)(ws + (36u << 20));          // 4 MB
  __bf16* hb   = (__bf16*)(ws + (40u << 20));          // 32 MB
  float*  xinv = (float*)(ws + (72u << 20));           // 32 KB
  float*  minv = (float*)(ws + (72u << 20) + 32768u);  // 32 KB

  // 1) Norms + bf16 casts.
  rownorm_cvt_kernel<<<R, 256, 0, stream>>>(x, xb, xinv, C);
  rownorm_cvt_kernel<<<M, 256, 0, stream>>>(ltm, mb, minv, C);
  cvt_bf16_kernel<<<512, 256, 0, stream>>>(W1, w1b, C * H);
  cvt_bf16_kernel<<<512, 256, 0, stream>>>(W2, w2b, H * C);

  // 2) MLP: h = gelu(x@W1+b1) -> bf16; mem_out = h@W2+b2 -> d_out (fp32).
  dim3 g1(H / 128, R / 128);
  gemm_bf16_wmma<0><<<g1, 256, 0, stream>>>(xb, w1b, b1, (void*)hb, R, H, C);
  dim3 g2(C / 128, R / 128);
  gemm_bf16_wmma<1><<<g2, 256, 0, stream>>>(hb, w2b, b2, (void*)out, R, C, H);

  // 3) Fused scores + top-16 + softmax + gather, accumulated into d_out.
  scores_topk_kernel<<<R / 128, 256, SC_LDS_SIZE, stream>>>(xb, mb, xinv, minv,
                                                            ltm, out, M);
}
